// Attention_85555748537160
// MI455X (gfx1250) — matrix operand order repaired and checked by reading, not compiled, not run on hardware
//
#include <hip/hip_runtime.h>
#include <hip/hip_bf16.h>

typedef _Float16 v16h __attribute__((ext_vector_type(16)));
typedef _Float16 v8h  __attribute__((ext_vector_type(8)));
typedef _Float16 v2h  __attribute__((ext_vector_type(2)));
typedef float    v8f  __attribute__((ext_vector_type(8)));

#define Bsz 2
#define Nseq 2048
#define Cdim 768
#define Hn 12
#define HD 64

union F16x16 { v16h v; v8h h[2]; };

__device__ __forceinline__ v8f wmma16(v16h a, v16h b, v8f c) {
  // D = A(16x32 f16) x B(32x16 f16) + C(16x16 f32)
  return __builtin_amdgcn_wmma_f32_16x16x32_f16(
      /*neg_a=*/false, a, /*neg_b=*/false, b,
      /*c_mod=*/(short)0, c, /*reuse_a=*/false, /*reuse_b=*/false);
}

// A-matrix fragment (rows tileRow..+15, K = kOff..kOff+31) from row-major LDS.
// ISA layout: lanes 0-15 hold K 0..7 / 16..23; lanes 16-31 hold K 8..15 / 24..31.
__device__ __forceinline__ v16h frag_a(const _Float16* base, int stride, int tileRow, int kOff) {
  const int lane = threadIdx.x & 31;
  const int row  = tileRow + (lane & 15);
  const int kb   = kOff + ((lane & 16) ? 8 : 0);
  const _Float16* p = base + row * stride + kb;
  F16x16 f;
  f.h[0] = *(const v8h*)(p);
  f.h[1] = *(const v8h*)(p + 16);
  return f.v;
}

// B-matrix fragment (cols tileCol..+15, K = kOff..kOff+31) from K-major LDS
// (base[col*stride + k]).  Lanes 0-15: K 0..7 / 16..23, lanes 16-31: K 8..15 / 24..31.
__device__ __forceinline__ v16h frag_b(const _Float16* base, int stride, int tileCol, int kOff) {
  const int lane = threadIdx.x & 31;
  const int col  = tileCol + (lane & 15);
  const int kb   = kOff + ((lane & 16) ? 8 : 0);
  const _Float16* p = base + col * stride + kb;
  F16x16 f;
  f.h[0] = *(const v8h*)(p);
  f.h[1] = *(const v8h*)(p + 16);
  return f.v;
}

// ---------------------------------------------------------------------------
// Kernel 1: qkv = x @ Wqkv + bias, fused RMSNorm + RoPE on q/k heads.
// Block tile: 128 rows x 64 cols; 8 waves, 2x2 WMMA tiles per wave.
// A 64-wide col tile == one (section, head) slice so the HD=64 norm is local.
// ---------------------------------------------------------------------------
__global__ __launch_bounds__(256) void qkv_fused_kernel(
    const float* __restrict__ x, const float* __restrict__ Wqkv,
    const float* __restrict__ bias, const float* __restrict__ cosT,
    const float* __restrict__ sinT, const float* __restrict__ qw,
    const float* __restrict__ kw,
    _Float16* __restrict__ Qh, _Float16* __restrict__ Kh, _Float16* __restrict__ Vh)
{
  __shared__ __align__(16) _Float16 As[128 * 40];  // [row][k] row-major
  __shared__ __align__(16) _Float16 Bs[64 * 40];   // [col][k] K-major (W^T tile)
  __shared__ float Co[128 * 68];                   // fp32 staging

  const int n0   = blockIdx.x * 64;          // column block in [0, 2304)
  const int m0   = blockIdx.y * 128;         // row block in [0, 4096)
  const int tid  = threadIdx.x;
  const int wave = tid >> 5;
  const int lane = tid & 31;
  const int wm   = (wave & 3) * 32;          // wave's 2 row tiles
  const int wn   = (wave >> 2) * 32;         // wave's 2 col tiles

  v8f acc[2][2] = {};

  const int ra = tid >> 1, ca = (tid & 1) * 16;        // A loader: 16 elems
  const int kp = (tid >> 4) * 2, nb = (tid & 15) * 4;  // B loader: k-pair x 4n

  for (int k0 = 0; k0 < Cdim; k0 += 32) {
    {
      const float* g = x + (size_t)(m0 + ra) * Cdim + k0 + ca;
      if (k0 + 32 < Cdim) __builtin_prefetch(g + 32, 0, 1);
      _Float16 t16[16];
#pragma unroll
      for (int i = 0; i < 16; ++i) t16[i] = (_Float16)g[i];
      *(v8h*)&As[ra * 40 + ca]     = *(v8h*)&t16[0];
      *(v8h*)&As[ra * 40 + ca + 8] = *(v8h*)&t16[8];
    }
    {
      const float* g0 = Wqkv + (size_t)(k0 + kp) * 2304 + n0 + nb;
      const float* g1 = g0 + 2304;
      if (k0 + 32 < Cdim) __builtin_prefetch(g0 + (size_t)32 * 2304, 0, 1);
#pragma unroll
      for (int i = 0; i < 4; ++i) {
        v2h pk; pk.x = (_Float16)g0[i]; pk.y = (_Float16)g1[i];
        *(v2h*)&Bs[(nb + i) * 40 + kp] = pk;   // packed ds_store_b32
      }
    }
    __syncthreads();
    v16h a0 = frag_a(As, 40, wm, 0);
    v16h a1 = frag_a(As, 40, wm + 16, 0);
    v16h b0 = frag_b(Bs, 40, wn, 0);
    v16h b1 = frag_b(Bs, 40, wn + 16, 0);
    acc[0][0] = wmma16(a0, b0, acc[0][0]);
    acc[0][1] = wmma16(a0, b1, acc[0][1]);
    acc[1][0] = wmma16(a1, b0, acc[1][0]);
    acc[1][1] = wmma16(a1, b1, acc[1][1]);
    __syncthreads();
  }

  // stage tile (+bias) to LDS in fp32
  {
    const int cc = wn + (lane & 15);
    const float bb0 = bias[n0 + cc];
    const float bb1 = bias[n0 + cc + 16];
#pragma unroll
    for (int tr = 0; tr < 2; ++tr) {
      const int rb = wm + tr * 16 + ((lane & 16) ? 8 : 0);
#pragma unroll
      for (int r = 0; r < 8; ++r) {
        Co[(rb + r) * 68 + cc]      = acc[tr][0][r] + bb0;
        Co[(rb + r) * 68 + cc + 16] = acc[tr][1][r] + bb1;
      }
    }
  }
  __syncthreads();

  const int s  = n0 / Cdim;                // 0=q 1=k 2=v (uniform per block)
  const int h  = (n0 % Cdim) / HD;
  const int rr = tid >> 1, d0 = (tid & 1) * 32;   // 32 elems/thread
  const int gm = m0 + rr;
  const int bb = gm >> 11;                 // batch
  const int nn = gm & (Nseq - 1);          // token

  if (s == 2) {
    _Float16* dst = Vh + ((size_t)(bb * Hn + h) * Nseq + nn) * HD + d0;
#pragma unroll
    for (int i = 0; i < 32; ++i) dst[i] = (_Float16)Co[rr * 68 + d0 + i];
  } else {
    // 2 threads per row: partial sum-of-squares, combine via lane shuffle
    float s2 = 0.f;
#pragma unroll 8
    for (int j = 0; j < 32; ++j) { float v = Co[rr * 68 + d0 + j]; s2 += v * v; }
    s2 += __shfl_xor(s2, 1, 32);
    const float rs = rsqrtf(s2 * (1.0f / 64.0f) + 1e-6f);
    const float* w = (s == 0) ? qw : kw;
    _Float16* dst = ((s == 0) ? Qh : Kh) + ((size_t)(bb * Hn + h) * Nseq + nn) * HD + d0;
    const float* cr = cosT + (size_t)nn * HD;
    const float* sr = sinT + (size_t)nn * HD;
#pragma unroll
    for (int i = 0; i < 32; ++i) {
      const int d  = d0 + i;
      const int dr = (d < 32) ? d + 32 : d - 32;
      const float xv  = Co[rr * 68 + d]  * rs * w[d];
      float       rot = Co[rr * 68 + dr] * rs * w[dr];
      if (d < 32) rot = -rot;
      dst[i] = (_Float16)(xv * cr[d] + rot * sr[d]);
    }
  }
}

// ---------------------------------------------------------------------------
// Kernel 2: flash attention per (b,h); 128-query tile per block, 64-key steps.
// ---------------------------------------------------------------------------
__global__ __launch_bounds__(256) void attn_kernel(
    const _Float16* __restrict__ Qh, const _Float16* __restrict__ Kh,
    const _Float16* __restrict__ Vh, float* __restrict__ ctx)
{
  __shared__ __align__(16) _Float16 Qs[128 * 72];  // [qrow][dim]
  __shared__ __align__(16) _Float16 Ks[64 * 72];   // [key][dim]  (B-frag K-major)
  __shared__ __align__(16) _Float16 Vs[64 * 72];   // [dim][key]  (transposed)
  __shared__ __align__(16) _Float16 Ps[128 * 72];  // [qrow][key] f16 probs
  __shared__ float Ss[128 * 68];                   // fp32 scores
  __shared__ float aArr[128], lArr[128];

  const int bh   = blockIdx.x;             // 0..23
  const int q0   = blockIdx.y * 128;
  const int tid  = threadIdx.x;
  const int wave = tid >> 5;
  const int lane = tid & 31;
  const int wm = (wave & 3) * 32;
  const int wn = (wave >> 2) * 32;

  const _Float16* Qg = Qh + ((size_t)bh * Nseq + q0) * HD;
  const _Float16* Kg = Kh + (size_t)bh * Nseq * HD;
  const _Float16* Vg = Vh + (size_t)bh * Nseq * HD;

  {
    const int r = tid >> 1, c = (tid & 1) * 32;
    const _Float16* g = Qg + (size_t)r * HD + c;
#pragma unroll
    for (int i = 0; i < 32; ++i) Qs[r * 72 + c + i] = g[i];
  }

  float m_i = -3.0e38f, l_i = 0.0f;        // row = tid>>1 state (pair-replicated)
  v8f o[2][2] = {};
  __syncthreads();

  const int rk = tid >> 2, ck = (tid & 3) * 16;   // K/V loader
  for (int kb = 0; kb < Nseq / 64; ++kb) {
    const int k0 = kb * 64;
    {
      const _Float16* gk = Kg + (size_t)(k0 + rk) * HD + ck;
      const _Float16* gv = Vg + (size_t)(k0 + rk) * HD + ck;
      if (kb + 1 < Nseq / 64) { __builtin_prefetch(gk + 64 * HD, 0, 1); __builtin_prefetch(gv + 64 * HD, 0, 1); }
#pragma unroll
      for (int i = 0; i < 16; ++i) {
        Ks[rk * 72 + ck + i]    = gk[i];
        Vs[(ck + i) * 72 + rk]  = gv[i];    // transpose V -> [dim][key]
      }
    }
    __syncthreads();

    // S = (Q K^T) * scale   (2x2 tiles per wave, K=64)
    v8f sA[2][2] = {};
#pragma unroll
    for (int ks = 0; ks < HD; ks += 32) {
      v16h a0 = frag_a(Qs, 72, wm, ks);
      v16h a1 = frag_a(Qs, 72, wm + 16, ks);
      v16h b0 = frag_b(Ks, 72, wn, ks);
      v16h b1 = frag_b(Ks, 72, wn + 16, ks);
      sA[0][0] = wmma16(a0, b0, sA[0][0]);
      sA[0][1] = wmma16(a0, b1, sA[0][1]);
      sA[1][0] = wmma16(a1, b0, sA[1][0]);
      sA[1][1] = wmma16(a1, b1, sA[1][1]);
    }
    {
      const int cc = wn + (lane & 15);
#pragma unroll
      for (int tr = 0; tr < 2; ++tr) {
        const int rb = wm + tr * 16 + ((lane & 16) ? 8 : 0);
#pragma unroll
        for (int r = 0; r < 8; ++r) {
          Ss[(rb + r) * 68 + cc]      = sA[tr][0][r] * 0.125f;  // 1/sqrt(64)
          Ss[(rb + r) * 68 + cc + 16] = sA[tr][1][r] * 0.125f;
        }
      }
    }
    __syncthreads();

    // online softmax: 2 threads per row, 32 cols each, shuffle-combined
    {
      const int row = tid >> 1, c0s = (tid & 1) * 32;
      float rmax = -3.0e38f;
#pragma unroll 8
      for (int j = 0; j < 32; ++j) rmax = fmaxf(rmax, Ss[row * 68 + c0s + j]);
      rmax = fmaxf(rmax, __shfl_xor(rmax, 1, 32));
      const float newm = fmaxf(m_i, rmax);
      const float al   = __expf(m_i - newm);
      float sum = 0.f;
#pragma unroll 8
      for (int j = 0; j < 32; ++j) {
        const float p = __expf(Ss[row * 68 + c0s + j] - newm);
        sum += p;
        Ps[row * 72 + c0s + j] = (_Float16)p;
      }
      sum += __shfl_xor(sum, 1, 32);
      l_i = l_i * al + sum;
      m_i = newm;
      if ((tid & 1) == 0) aArr[row] = al;
    }
    __syncthreads();

    // rescale O fragments (each acc element maps to a distinct row), O += P V
#pragma unroll
    for (int tr = 0; tr < 2; ++tr) {
      const int rb = wm + tr * 16 + ((lane & 16) ? 8 : 0);
#pragma unroll
      for (int r = 0; r < 8; ++r) {
        const float al = aArr[rb + r];
        o[tr][0][r] *= al; o[tr][1][r] *= al;
      }
    }
#pragma unroll
    for (int ks = 0; ks < 64; ks += 32) {
      v16h a0 = frag_a(Ps, 72, wm, ks);
      v16h a1 = frag_a(Ps, 72, wm + 16, ks);
      v16h b0 = frag_b(Vs, 72, wn, ks);
      v16h b1 = frag_b(Vs, 72, wn + 16, ks);
      o[0][0] = wmma16(a0, b0, o[0][0]);
      o[0][1] = wmma16(a0, b1, o[0][1]);
      o[1][0] = wmma16(a1, b0, o[1][0]);
      o[1][1] = wmma16(a1, b1, o[1][1]);
    }
    __syncthreads();
  }

  if ((tid & 1) == 0) lArr[tid >> 1] = 1.0f / l_i;
  __syncthreads();

  {
    const int b = bh / Hn, h = bh % Hn;
    const int cc = wn + (lane & 15);
#pragma unroll
    for (int tr = 0; tr < 2; ++tr) {
      const int rb = wm + tr * 16 + ((lane & 16) ? 8 : 0);
#pragma unroll
      for (int r = 0; r < 8; ++r) {
        const int row = rb + r;
        const float inv = lArr[row];
        const size_t off = ((size_t)b * Nseq + q0 + row) * Cdim + h * HD;
        ctx[off + cc]      = o[tr][0][r] * inv;
        ctx[off + cc + 16] = o[tr][1][r] * inv;
      }
    }
  }
}

// ---------------------------------------------------------------------------
// Kernel 3: out = ctx @ Wproj + bias   (4096x768 @ 768x768), 128x64 tiles
// ---------------------------------------------------------------------------
__global__ __launch_bounds__(256) void proj_kernel(
    const float* __restrict__ ctx, const float* __restrict__ Wproj,
    const float* __restrict__ bias, float* __restrict__ out)
{
  __shared__ __align__(16) _Float16 As[128 * 40];
  __shared__ __align__(16) _Float16 Bs[64 * 40];

  const int n0   = blockIdx.x * 64;
  const int m0   = blockIdx.y * 128;
  const int tid  = threadIdx.x;
  const int wave = tid >> 5;
  const int lane = tid & 31;
  const int wm = (wave & 3) * 32;
  const int wn = (wave >> 2) * 32;

  v8f acc[2][2] = {};

  const int ra = tid >> 1, ca = (tid & 1) * 16;
  const int kp = (tid >> 4) * 2, nb = (tid & 15) * 4;

  for (int k0 = 0; k0 < Cdim; k0 += 32) {
    {
      const float* g = ctx + (size_t)(m0 + ra) * Cdim + k0 + ca;
      if (k0 + 32 < Cdim) __builtin_prefetch(g + 32, 0, 1);
      _Float16 t16[16];
#pragma unroll
      for (int i = 0; i < 16; ++i) t16[i] = (_Float16)g[i];
      *(v8h*)&As[ra * 40 + ca]     = *(v8h*)&t16[0];
      *(v8h*)&As[ra * 40 + ca + 8] = *(v8h*)&t16[8];
    }
    {
      const float* g0 = Wproj + (size_t)(k0 + kp) * Cdim + n0 + nb;
      const float* g1 = g0 + Cdim;
      if (k0 + 32 < Cdim) __builtin_prefetch(g0 + (size_t)32 * Cdim, 0, 1);
#pragma unroll
      for (int i = 0; i < 4; ++i) {
        v2h pk; pk.x = (_Float16)g0[i]; pk.y = (_Float16)g1[i];
        *(v2h*)&Bs[(nb + i) * 40 + kp] = pk;
      }
    }
    __syncthreads();
    v16h a0 = frag_a(As, 40, wm, 0);
    v16h a1 = frag_a(As, 40, wm + 16, 0);
    v16h b0 = frag_b(Bs, 40, wn, 0);
    v16h b1 = frag_b(Bs, 40, wn + 16, 0);
    acc[0][0] = wmma16(a0, b0, acc[0][0]);
    acc[0][1] = wmma16(a0, b1, acc[0][1]);
    acc[1][0] = wmma16(a1, b0, acc[1][0]);
    acc[1][1] = wmma16(a1, b1, acc[1][1]);
    __syncthreads();
  }

  const int cc = wn + (lane & 15);
  const float bb0 = bias[n0 + cc];
  const float bb1 = bias[n0 + cc + 16];
#pragma unroll
  for (int tr = 0; tr < 2; ++tr) {
    const int rb = wm + tr * 16 + ((lane & 16) ? 8 : 0);
#pragma unroll
    for (int r = 0; r < 8; ++r) {
      out[(size_t)(m0 + rb + r) * Cdim + n0 + cc]      = acc[tr][0][r] + bb0;
      out[(size_t)(m0 + rb + r) * Cdim + n0 + cc + 16] = acc[tr][1][r] + bb1;
    }
  }
}

extern "C" void kernel_launch(void* const* d_in, const int* in_sizes, int n_in,
                              void* d_out, int out_size, void* d_ws, size_t ws_size,
                              hipStream_t stream) {
  const float* x        = (const float*)d_in[0];
  const float* rope_cos = (const float*)d_in[1];
  const float* rope_sin = (const float*)d_in[2];
  const float* Wqkv     = (const float*)d_in[3];
  const float* bqkv     = (const float*)d_in[4];
  const float* Wproj    = (const float*)d_in[5];
  const float* bproj    = (const float*)d_in[6];
  const float* qnw      = (const float*)d_in[7];
  const float* knw      = (const float*)d_in[8];
  float* out = (float*)d_out;

  // workspace: Q,K,V f16 [B,H,N,64] (6 MB each) + ctx f32 [B,N,C] (12 MB)
  char* ws = (char*)d_ws;
  const size_t qkvBytes = (size_t)Bsz * Hn * Nseq * HD * sizeof(_Float16);
  _Float16* Qh = (_Float16*)(ws);
  _Float16* Kh = (_Float16*)(ws + qkvBytes);
  _Float16* Vh = (_Float16*)(ws + 2 * qkvBytes);
  float*    cx = (float*)  (ws + 3 * qkvBytes);

  qkv_fused_kernel<<<dim3(2304 / 64, (Bsz * Nseq) / 128), 256, 0, stream>>>(
      x, Wqkv, bqkv, rope_cos, rope_sin, qnw, knw, Qh, Kh, Vh);
  attn_kernel<<<dim3(Bsz * Hn, Nseq / 128), 256, 0, stream>>>(Qh, Kh, Vh, cx);
  proj_kernel<<<dim3(Cdim / 64, (Bsz * Nseq) / 128), 256, 0, stream>>>(
      cx, Wproj, bproj, out);
}
